// DownLayer_57226144252765
// MI455X (gfx1250) — matrix operand order repaired and checked by reading, not compiled, not run on hardware
//
#include <hip/hip_runtime.h>
#include <hip/hip_bf16.h>

typedef __attribute__((ext_vector_type(16))) _Float16 v16h;
typedef __attribute__((ext_vector_type(8)))  float    v8f;

#define BATCH 64
#define NTOK  3136
#define CIN   64
#define COUT  128
#define HH    56
#define WW    56
#define HW    (HH*WW)
#define NGRID 196
#define NADA  2940
#define SAMP  980
#define NOUT  1176          // NGRID + SAMP
#define KTOT  576           // 9 taps * 64 cin
#define NKK   (KTOT/32)     // 18 K-chunks of 32

// normalized 1D gaussian, ks=9, sigma=2 (separable form of the 9x9 kernel)
__device__ __constant__ float G9[9] = {
    0.02763055f, 0.06628157f, 0.12383154f, 0.18017382f, 0.20416369f,
    0.18017382f, 0.12383154f, 0.06628157f, 0.02763055f};

// ---------------- scatter tokens into (B,H,W,65) grid ----------------
__global__ __launch_bounds__(256) void k_scatter(const float* __restrict__ x,
                                                 const float* __restrict__ pos,
                                                 float* __restrict__ grid65) {
    int t = blockIdx.x * 256 + threadIdx.x;           // token id over B*N
    if (t >= BATCH * NTOK) return;
    int b = t / NTOK;
    float lx = fminf(fmaxf(pos[(size_t)t * 2 + 0], 0.f), 1.f);
    float ly = fminf(fmaxf(pos[(size_t)t * 2 + 1], 0.f), 1.f);
    int px = (int)rintf(lx * (WW - 1));               // rint = round-half-even (matches jnp.round)
    int py = (int)rintf(ly * (HH - 1));
    size_t dst = ((size_t)b * HW + (size_t)py * WW + px) * 65;
    const float* src = x + (size_t)t * CIN;
    for (int c = 0; c < CIN; ++c) atomicAdd(&grid65[dst + c], src[c]);
    atomicAdd(&grid65[dst + CIN], 1.0f);
}

// ---------------- normalize by count, binarize mask (in place) ----------------
__global__ __launch_bounds__(256) void k_norm(float* __restrict__ grid65) {
    int p = blockIdx.x * 256 + threadIdx.x;           // pixel over B*HW
    if (p >= BATCH * HW) return;
    size_t base = (size_t)p * 65;
    float m = grid65[base + CIN];
    float inv = 1.0f / (m + 1e-6f);
    float mb = (m > 0.f) ? 1.0f : 0.0f;
    for (int c = 0; c < CIN; ++c) grid65[base + c] = grid65[base + c] * inv * mb;
    grid65[base + CIN] = mb;
}

// ---------------- separable 9-tap blur, horizontal then vertical ----------------
__global__ __launch_bounds__(256) void k_blur_h(const float* __restrict__ in,
                                                float* __restrict__ out) {
    size_t i = (size_t)blockIdx.x * 256 + threadIdx.x;  // over B*HW*65
    if (i >= (size_t)BATCH * HW * 65) return;
    int ch = (int)(i % 65);
    size_t pix = i / 65;
    int xx = (int)(pix % WW);
    size_t rowbase = (pix - xx) * 65 + ch;
    float acc = 0.f;
    #pragma unroll
    for (int d = -4; d <= 4; ++d) {
        int xs = xx + d;
        if (xs >= 0 && xs < WW) acc += G9[d + 4] * in[rowbase + (size_t)xs * 65];
    }
    out[i] = acc;
}

__global__ __launch_bounds__(256) void k_blur_v(const float* __restrict__ in,
                                                float* __restrict__ out) {
    size_t i = (size_t)blockIdx.x * 256 + threadIdx.x;
    if (i >= (size_t)BATCH * HW * 65) return;
    int ch = (int)(i % 65);
    size_t pix = i / 65;
    int xx = (int)(pix % WW);
    int yy = (int)((pix / WW) % HH);
    size_t imgbase = (pix - (size_t)yy * WW - xx) * 65;
    float acc = 0.f;
    #pragma unroll
    for (int d = -4; d <= 4; ++d) {
        int ys = yy + d;
        if (ys >= 0 && ys < HH)
            acc += G9[d + 4] * in[imgbase + ((size_t)ys * WW + xx) * 65 + ch];
    }
    out[i] = acc;
}

// ---------------- hole fill: feat += (1-mask) * blur_feat/blur_mask ----------------
__global__ __launch_bounds__(256) void k_hole(const float* __restrict__ blur65,
                                              float* __restrict__ grid65) {
    size_t i = (size_t)blockIdx.x * 256 + threadIdx.x;  // over B*HW*64
    if (i >= (size_t)BATCH * HW * CIN) return;
    int ch = (int)(i % CIN);
    size_t pix = i / CIN;
    size_t base = pix * 65;
    float m  = grid65[base + CIN];
    float bm = blur65[base + CIN];
    float fi = blur65[base + ch] / (bm + 1e-6f) * ((bm > 0.f) ? 1.f : 0.f);
    grid65[base + ch] += (1.f - m) * fi;
}

// ---------------- pre-pack conv weights into B-fragment order (f16) ----------------
// wpack[w(8)][kk(18)][lane(32)][e(16)]: lane L -> co = w*16 + (L&15),
// K = kk*32 + (e&8)*2 + (L>>4)*8 + (e&7). Each lane's fragment = 32 contiguous bytes.
__global__ __launch_bounds__(256) void k_pack_w(const float* __restrict__ w,
                                                _Float16* __restrict__ wpack) {
    int i = blockIdx.x * 256 + threadIdx.x;           // over 8*18*32*16 = 73728
    if (i >= 8 * NKK * 32 * 16) return;
    int e    = i & 15;
    int lane = (i >> 4) & 31;
    int kk   = (i >> 9) % NKK;
    int wv   = i / (NKK * 512);
    int K  = kk * 32 + (e & 8) * 2 + (lane >> 4) * 8 + (e & 7);
    int co = wv * 16 + (lane & 15);
    wpack[i] = (_Float16)w[(size_t)K * COUT + co];    // conv_w (3,3,64,128) HWIO flat
}

// ---------------- 3x3 conv 64->128 as implicit GEMM via WMMA f16 ----------------
// block: 256 thr (8 waves) = (b, 16 spatial rows) x 128 out channels.
// A staged in LDS in *fragment order* so each lane reads its v16h contiguously;
// B fragments come pre-packed from global. K loop fully unrolled -> 18 wmma.
__global__ __launch_bounds__(256) void k_conv_wmma(const float* __restrict__ grid65,
                                                   const _Float16* __restrict__ wpack,
                                                   const float* __restrict__ bias,
                                                   float* __restrict__ fmap) {
    __shared__ __align__(32) _Float16 a_frag[NKK][32][16];   // 18 KB
    int blk = blockIdx.x;
    int b  = blk / (HW / 16);
    int mt = blk % (HW / 16);
    int p0 = mt * 16;

    // Cooperative A staging, source-ordered (coalesced along cin), writing the
    // inverse of the 16-bit A-matrix 16x32 VGPR permutation:
    //   km<16 : half=km>>3, e=km&7 ; km>=16: t=km-16, half=t>>3, e=8+(t&7)
    for (int i = threadIdx.x; i < 16 * KTOT; i += 256) {
        int row = i / KTOT, k = i % KTOT;
        int tap = k / CIN, cin = k % CIN;
        int p = p0 + row;
        int y = p / WW, xx = p % WW;
        int yy = y + tap / 3 - 1, xx2 = xx + tap % 3 - 1;
        float v = 0.f;
        if (yy >= 0 && yy < HH && xx2 >= 0 && xx2 < WW)
            v = grid65[((size_t)b * HW + (size_t)yy * WW + xx2) * 65 + cin];
        int kk = k >> 5, km = k & 31;
        int half, e;
        if (km < 16) { half = km >> 3; e = km & 7; }
        else         { int t2 = km - 16; half = t2 >> 3; e = 8 + (t2 & 7); }
        a_frag[kk][half * 16 + row][e] = (_Float16)v;
    }
    __syncthreads();

    int wave = threadIdx.x >> 5;       // N-tile (0..7) -> out channels wave*16..+15
    int lane = threadIdx.x & 31;
    int half = lane >> 4, ln = lane & 15;
    int co = wave * 16 + ln;
    const v16h* bsrc = (const v16h*)(wpack + ((size_t)wave * NKK * 32 + lane) * 16);

    v8f acc = {};
    #pragma unroll
    for (int kk = 0; kk < NKK; ++kk) {
        v16h af = *(const v16h*)&a_frag[kk][lane][0];   // 2x ds_load_b128
        v16h bf = bsrc[(size_t)kk * 32];                // 2x global_load_b128, coalesced
        acc = __builtin_amdgcn_wmma_f32_16x16x32_f16(
            false, af, false, bf, (short)0, acc, false, false);
    }
    // C/D layout: vgpr v -> M = v + 8*half, N = ln
    float bv = bias[co];
    #pragma unroll
    for (int v = 0; v < 8; ++v) {
        int p = p0 + v + 8 * half;
        fmap[((size_t)b * HW + p) * COUT + co] = acc[v] + bv;
    }
}

// ---------------- helpers: bilinear sample + block-wide reduce (128 thr) ----------------
__device__ inline float bsample(const float* __restrict__ fm, int H_, int W_, int C_,
                                float lx, float ly, int c) {
    float gx = lx * 2.f - 1.f, gy = ly * 2.f - 1.f;
    float ix = ((gx + 1.f) * W_ - 1.f) * 0.5f;
    float iy = ((gy + 1.f) * H_ - 1.f) * 0.5f;
    float x0 = floorf(ix), y0 = floorf(iy);
    float wx1 = ix - x0, wy1 = iy - y0;
    float acc = 0.f;
    #pragma unroll
    for (int dy = 0; dy < 2; ++dy)
        #pragma unroll
        for (int dx = 0; dx < 2; ++dx) {
            float xc = x0 + dx, yc = y0 + dy;
            float wgt = (dx ? wx1 : 1.f - wx1) * (dy ? wy1 : 1.f - wy1);
            bool valid = (xc >= 0.f) && (xc <= (float)(W_ - 1)) &&
                         (yc >= 0.f) && (yc <= (float)(H_ - 1));
            int xi = (int)fminf(fmaxf(xc, 0.f), (float)(W_ - 1));
            int yi = (int)fminf(fmaxf(yc, 0.f), (float)(H_ - 1));
            acc += fm[((size_t)yi * W_ + xi) * C_ + c] * (valid ? wgt : 0.f);
        }
    return acc;
}

__device__ inline float block_sum128(float v, float* red, int t) {
    red[t] = v; __syncthreads();
    for (int s = 64; s > 0; s >>= 1) { if (t < s) red[t] += red[t + s]; __syncthreads(); }
    float r = red[0]; __syncthreads();
    return r;
}

// ---------------- confidence: bilinear + LN + dot(conf_w), one block per token ----------------
__global__ __launch_bounds__(128) void k_conf(const float* __restrict__ fmap,
                                              const float* __restrict__ pos,
                                              const float* __restrict__ nw,
                                              const float* __restrict__ nb,
                                              const float* __restrict__ cw,
                                              const float* __restrict__ cb,
                                              float* __restrict__ conf) {
    __shared__ float red[128];
    int tok = blockIdx.x;                       // over B*N
    int b = tok / NTOK;
    int c = threadIdx.x;
    float lx = pos[(size_t)tok * 2 + 0], ly = pos[(size_t)tok * 2 + 1];
    float v = bsample(fmap + (size_t)b * HW * COUT, HH, WW, COUT, lx, ly, c);
    float mu = block_sum128(v, red, c) * (1.f / COUT);
    float d = v - mu;
    float var = block_sum128(d * d, red, c) * (1.f / COUT);
    float xn = d * rsqrtf(var + 1e-5f) * nw[c] + nb[c];
    float cv = block_sum128(xn * cw[c], red, c) + cb[0];
    if (c == 0) conf[tok] = cv;
}

// ---------------- gumbel scores + descending bitonic top-k (sorted) ----------------
__device__ inline float hash_uniform(unsigned b, unsigned i) {
    // stateless hash -> (0,1). NOTE: not bit-identical to JAX threefry.
    unsigned s = b * 2940u + i + 0x9E3779B9u;
    s ^= s >> 16; s *= 0x7FEB352Du; s ^= s >> 15; s *= 0x846CA68Bu; s ^= s >> 16;
    return ((s >> 8) + 0.5f) * (1.0f / 16777216.0f);
}

__global__ __launch_bounds__(1024) void k_topk(const float* __restrict__ conf,
                                               int* __restrict__ idx) {
    __shared__ float key[4096];
    __shared__ int   vi[4096];
    int b = blockIdx.x;
    for (int i = threadIdx.x; i < 4096; i += 1024) {
        float s = -3.4e38f;
        if (i < NADA) {
            float u = hash_uniform(b, i);
            float gn = -logf(-logf(u + 1e-6f) + 1e-6f);     // gumbel
            s = conf[(size_t)b * NTOK + NGRID + i] + gn;    // T = 1
        }
        key[i] = s; vi[i] = i;
    }
    __syncthreads();
    for (int k = 2; k <= 4096; k <<= 1) {
        for (int j = k >> 1; j > 0; j >>= 1) {
            for (int i = threadIdx.x; i < 4096; i += 1024) {
                int p = i ^ j;
                if (p > i) {
                    bool desc = ((i & k) == 0);
                    float a = key[i], bb = key[p];
                    bool swp = desc ? (a < bb) : (a > bb);
                    if (swp) {
                        key[i] = bb; key[p] = a;
                        int t = vi[i]; vi[i] = vi[p]; vi[p] = t;
                    }
                }
            }
            __syncthreads();
        }
    }
    for (int i = threadIdx.x; i < SAMP; i += 1024) idx[b * SAMP + i] = vi[i];
}

// ---------------- assemble output: recompute LN sample for selected tokens + pos_embed ----------------
__global__ __launch_bounds__(128) void k_out(const float* __restrict__ fmap,
                                             const float* __restrict__ pos,
                                             const float* __restrict__ pe,
                                             const float* __restrict__ nw,
                                             const float* __restrict__ nb,
                                             const int* __restrict__ idx,
                                             float* __restrict__ xout,
                                             float* __restrict__ pout) {
    __shared__ float red[128];
    int blk = blockIdx.x;                       // over B*NOUT
    int b = blk / NOUT, ti = blk % NOUT;
    int c = threadIdx.x;
    int src = (ti < NGRID) ? ti : (NGRID + idx[b * SAMP + (ti - NGRID)]);
    size_t stok = (size_t)b * NTOK + src;
    float lx = pos[stok * 2 + 0], ly = pos[stok * 2 + 1];
    // bilinear + layernorm (recompute; xt never materialized)
    float v = bsample(fmap + (size_t)b * HW * COUT, HH, WW, COUT, lx, ly, c);
    float mu = block_sum128(v, red, c) * (1.f / COUT);
    float d = v - mu;
    float var = block_sum128(d * d, red, c) * (1.f / COUT);
    float xn = d * rsqrtf(var + 1e-5f) * nw[c] + nb[c];
    // add pos_embed bilinear sample (1,14,14,128 broadcast)
    float peb = bsample(pe, 14, 14, COUT, lx, ly, c);
    xout[((size_t)b * NOUT + ti) * COUT + c] = xn + peb;
    if (c < 2) pout[((size_t)b * NOUT + ti) * 2 + c] = pos[stok * 2 + c];
}

// ---------------- launch ----------------
extern "C" void kernel_launch(void* const* d_in, const int* in_sizes, int n_in,
                              void* d_out, int out_size, void* d_ws, size_t ws_size,
                              hipStream_t stream) {
    const float* x     = (const float*)d_in[0];
    const float* pos   = (const float*)d_in[1];
    const float* pe    = (const float*)d_in[2];
    const float* convw = (const float*)d_in[3];
    const float* convb = (const float*)d_in[4];
    const float* normw = (const float*)d_in[5];
    const float* normb = (const float*)d_in[6];
    const float* confw = (const float*)d_in[7];
    const float* confb = (const float*)d_in[8];

    const size_t n65  = (size_t)BATCH * HW * 65;    // 13,045,760
    const size_t n128 = (size_t)BATCH * HW * COUT;  // 25,690,112
    float* buf0 = (float*)d_ws;          // grid65 -> filled feat
    float* buf1 = buf0 + n65;            // h-blur temp
    float* buf2 = buf1 + n65;            // v-blur temp, then conv output (128ch)
    float* conf = buf2 + n128;           // B*N confidences
    int*   idxb = (int*)(conf + (size_t)BATCH * NTOK);
    _Float16* wpack = (_Float16*)(idxb + BATCH * SAMP);   // 73728 f16 = 147 KB

    float* xout = (float*)d_out;                               // (B,1176,128)
    float* pout = xout + (size_t)BATCH * NOUT * COUT;          // (B,1176,2)

    hipMemsetAsync(buf0, 0, n65 * sizeof(float), stream);

    k_pack_w <<<(8 * NKK * 32 * 16 + 255) / 256, 256, 0, stream>>>(convw, wpack);
    k_scatter<<<(BATCH * NTOK + 255) / 256, 256, 0, stream>>>(x, pos, buf0);
    k_norm   <<<(BATCH * HW + 255) / 256,  256, 0, stream>>>(buf0);
    k_blur_h <<<(int)((n65 + 255) / 256),  256, 0, stream>>>(buf0, buf1);
    k_blur_v <<<(int)((n65 + 255) / 256),  256, 0, stream>>>(buf1, buf2);
    k_hole   <<<(int)(((size_t)BATCH * HW * CIN + 255) / 256), 256, 0, stream>>>(buf2, buf0);
    k_conv_wmma<<<BATCH * (HW / 16), 256, 0, stream>>>(buf0, wpack, convb, buf2);
    k_conf   <<<BATCH * NTOK, 128, 0, stream>>>(buf2, pos, normw, normb, confw, confb, conf);
    k_topk   <<<BATCH, 1024, 0, stream>>>(conf, idxb);
    k_out    <<<BATCH * NOUT, 128, 0, stream>>>(buf2, pos, pe, normw, normb, idxb, xout, pout);
}
